// STAttentionRecBlockt_66984309949093
// MI455X (gfx1250) — matrix operand order repaired and checked by reading, not compiled, not run on hardware
//
#include <hip/hip_runtime.h>
#include <hip/hip_bf16.h>
#include <math.h>

// ---------------------------------------------------------------------------
// ST attention block for MI455X (gfx1250, wave32, WMMA).
// bf16 end-to-end activations for GEMM operands, f32 accumulate/residual.
// All WMMA kernels use register-staged double-buffered LDS pipelines so
// global loads of tile k+1 overlap with v_wmma of tile k.
// ---------------------------------------------------------------------------

typedef __attribute__((ext_vector_type(16))) __bf16 v16bf;
typedef __attribute__((ext_vector_type(8)))  float  v8f;
typedef unsigned int u32;

#define C_DIM  256
#define T_DIM  400
#define V_DIM  25
#define L_DIM  10000      // T*V
#define S_DIM  3
#define CI_DIM 64
#define QKC    384        // 2*S*CI
#define ZC     768        // S*C
#define NB     4          // batch

__device__ __forceinline__ float lrelu(float x){ return x >= 0.f ? x : 0.1f*x; }

// A-matrix (16x32, 16-bit) per-lane K index pattern (ISA 7.12.2):
__device__ __forceinline__ int a_kidx(int j, int g){
    return (j < 4) ? (8*g + 2*j) : (16 + 8*g + 2*(j-4));
}
// B-matrix (32x16, 16-bit): same K pattern as A
__device__ __forceinline__ int b_kidx(int j, int g){
    return (j < 4) ? (8*g + 2*j) : (16 + 8*g + 2*(j-4));
}

// ---------------------------------------------------------------------------
__global__ void cvt_bf16_kernel(const float* __restrict__ src, __bf16* __restrict__ dst, int n){
    int i = blockIdx.x * blockDim.x + threadIdx.x;
    if (i < n) dst[i] = (__bf16)src[i];
}

// yb[n][c][l] = bf16(x[n][c][l] + pe[c][l])
__global__ void add_pe_kernel(const float* __restrict__ x, const float* __restrict__ pe,
                              __bf16* __restrict__ y, int n_total){
    int i = blockIdx.x * blockDim.x + threadIdx.x;
    if (i < n_total) y[i] = (__bf16)(x[i] + pe[i % (C_DIM * L_DIM)]);
}

// ---------------------------------------------------------------------------
// conv1x1 GEMM:  acc[o][l] = sum_c W[o][c] * In[n][c][l]
// mode 0: Out(bf16) = acc + bias[o]
// mode 1: Out       = lrelu(res + gamma[o]*(acc+bias) + beta[o]) (bf16 or f32)
// grid: (L/16, O/(64*OT), n). 128 thr = 4 waves; wave owns OT o-tiles.
// Double-buffered LDS; loads for chunk c+1 issued before wmma of chunk c.
// ---------------------------------------------------------------------------
template<int OT>
__global__ __launch_bounds__(128)
void conv_gemm_kernel(const __bf16* __restrict__ W, const float* __restrict__ bias,
                      const __bf16* __restrict__ In, void* __restrict__ Out,
                      int O, int K,
                      const float* __restrict__ res, const float* __restrict__ gamma,
                      const float* __restrict__ beta, int mode, int outBf16)
{
    constexpr int ROWS = 64 * OT;                 // o rows per block
    constexpr int AREG = OT * 8;                  // staged u32 per thread
    __shared__ u32    Asu[2][ROWS][17];           // 34 bf16 per row (padded)
    __shared__ __bf16 Bs[2][16][34];
    const int tid  = threadIdx.x;
    const int lane = tid & 31, wave = tid >> 5;
    const int l0    = blockIdx.x * 16;
    const int oBase = blockIdx.y * ROWS;
    const int n     = blockIdx.z;
    const long nKL  = (long)n * K * L_DIM;
    const int row = lane & 15, g = lane >> 4;
    const u32* __restrict__ Wu = (const u32*)W;   // 2 bf16 per u32

    u32    areg[AREG];
    __bf16 breg[4];

    auto loadRegs = [&](int k0){
        #pragma unroll
        for (int i = 0; i < AREG; ++i){
            int e = tid + i * 128; int r = e >> 4, j = e & 15;
            areg[i] = Wu[((long)(oBase + r) * K + k0) / 2 + j];
        }
        #pragma unroll
        for (int i = 0; i < 4; ++i){
            int e = tid + i * 128; int k = e >> 4, l = e & 15;
            breg[i] = In[nKL + (long)(k0 + k) * L_DIM + l0 + l];
        }
    };
    auto storeLds = [&](int p){
        #pragma unroll
        for (int i = 0; i < AREG; ++i){
            int e = tid + i * 128; int r = e >> 4, j = e & 15;
            Asu[p][r][j] = areg[i];
        }
        #pragma unroll
        for (int i = 0; i < 4; ++i){
            int e = tid + i * 128; int k = e >> 4, l = e & 15;
            Bs[p][l][k] = breg[i];
        }
    };

    v8f acc[OT];
    #pragma unroll
    for (int ot = 0; ot < OT; ++ot) acc[ot] = (v8f){};

    const int NC = K / 32;
    loadRegs(0);
    storeLds(0);
    int p = 0;
    for (int c = 0; c < NC; ++c){
        __syncthreads();
        if (c + 1 < NC) loadRegs((c + 1) * 32);   // async: overlaps wmma below
        const __bf16* As = (const __bf16*)&Asu[p][0][0];
        v16bf b;
        #pragma unroll
        for (int j = 0; j < 8; ++j){
            int kb = b_kidx(j, g);
            b[2*j] = Bs[p][row][kb];  b[2*j+1] = Bs[p][row][kb + 1];
        }
        #pragma unroll
        for (int ot = 0; ot < OT; ++ot){
            const __bf16* Arow = As + (long)((wave * OT + ot) * 16 + row) * 34;
            v16bf a;
            #pragma unroll
            for (int j = 0; j < 8; ++j){
                int ka = a_kidx(j, g);
                a[2*j] = Arow[ka];  a[2*j+1] = Arow[ka + 1];
            }
            acc[ot] = __builtin_amdgcn_wmma_f32_16x16x32_bf16(false, a, false, b,
                                                              (short)0, acc[ot], false, false);
        }
        if (c + 1 < NC) storeLds(p ^ 1);          // drains staged regs
        p ^= 1;
    }

    const int l = l0 + row;                               // D: lane -> N(=l)
    #pragma unroll
    for (int ot = 0; ot < OT; ++ot){
        #pragma unroll
        for (int r = 0; r < 8; ++r){                      // VGPR r -> M = r+8g
            int o = oBase + (wave * OT + ot) * 16 + r + 8*g;
            long oidx = (long)n * O * L_DIM + (long)o * L_DIM + l;
            float v = acc[ot][r] + bias[o];
            if (mode == 1){
                v = res[oidx] + gamma[o]*v + beta[o];
                v = lrelu(v);
            }
            if (outBf16) ((__bf16*)Out)[oidx] = (__bf16)v;
            else         ((float*)Out)[oidx]  = v;
        }
    }
}

// ---------------------------------------------------------------------------
// Spatial attention scores (tiny: 25x25 per (n,s), reduce 64*400) — VALU.
// ---------------------------------------------------------------------------
__global__ __launch_bounds__(256)
void spatial_att_kernel(const __bf16* __restrict__ qk, const float* __restrict__ alphas,
                        const float* __restrict__ att0, float* __restrict__ att)
{
    __shared__ float red[256];
    const int u = blockIdx.x / V_DIM, w = blockIdx.x % V_DIM;
    const int n = blockIdx.y / S_DIM, s = blockIdx.y % S_DIM;
    const long qbase = ((long)n * QKC + s * CI_DIM) * L_DIM;
    const long kbase = qbase + (long)S_DIM * CI_DIM * L_DIM;
    float p = 0.f;
    for (int i = threadIdx.x; i < CI_DIM * T_DIM; i += 256){
        int c = i / T_DIM, t = i % T_DIM;
        long off = (long)c * L_DIM + t * V_DIM;
        p += (float)qk[qbase + off + u] * (float)qk[kbase + off + w];
    }
    red[threadIdx.x] = p;
    __syncthreads();
    for (int st = 128; st > 0; st >>= 1){
        if (threadIdx.x < st) red[threadIdx.x] += red[threadIdx.x + st];
        __syncthreads();
    }
    if (threadIdx.x == 0){
        float v = tanhf(red[0] * (1.f / (CI_DIM * T_DIM))) * alphas[s]
                + att0[(s * V_DIM + u) * V_DIM + w];
        att[(((long)n * S_DIM + s) * V_DIM + u) * V_DIM + w] = v;
    }
}

// ---------------------------------------------------------------------------
// z[n][s*C+c][t][w] = sum_v x[n][c][t][v] * att_s[n][s][v][w]  (V=25, VALU)
// ---------------------------------------------------------------------------
__global__ void spatial_apply_kernel(const float* __restrict__ x, const float* __restrict__ att,
                                     __bf16* __restrict__ z)
{
    long idx = (long)blockIdx.x * blockDim.x + threadIdx.x;
    if (idx >= (long)NB * ZC * L_DIM) return;
    int w = (int)(idx % V_DIM);
    long r = idx / V_DIM;
    int t = (int)(r % T_DIM); r /= T_DIM;
    int c = (int)(r % C_DIM); r /= C_DIM;
    int s = (int)(r % S_DIM); r /= S_DIM;
    int n = (int)r;
    const float* xrow = x + ((long)n * C_DIM + c) * L_DIM + t * V_DIM;
    const float* acol = att + (((long)n * S_DIM + s) * V_DIM) * V_DIM + w;
    float acc = 0.f;
    #pragma unroll
    for (int v = 0; v < V_DIM; ++v) acc += xrow[v] * acol[v * V_DIM];
    z[idx] = (__bf16)acc;
}

// ---------------------------------------------------------------------------
// Temporal attention scores (400x400 per (n,s), reduce 1600) — WMMA.
// grid: (T/16, T/16, n*S), 1 wave/block, double-buffered.
// ---------------------------------------------------------------------------
__global__ __launch_bounds__(32)
void temporal_att_kernel(const __bf16* __restrict__ qk, const float* __restrict__ alphat,
                         const float* __restrict__ att0, float* __restrict__ att)
{
    __shared__ __bf16 As[2][16][34];
    __shared__ __bf16 Bs[2][16][34];
    const int lane = threadIdx.x;
    const int q0 = blockIdx.x * 16, t0 = blockIdx.y * 16;
    const int n = blockIdx.z / S_DIM, s = blockIdx.z % S_DIM;
    const long qbase = ((long)n * QKC + s * CI_DIM) * L_DIM;
    const long kbase = qbase + (long)S_DIM * CI_DIM * L_DIM;
    const int row = lane & 15, g = lane >> 4;

    __bf16 armg[16], brmg[16];
    auto loadRegs = [&](int kk){
        #pragma unroll
        for (int i = 0; i < 16; ++i){
            int e = lane + i * 32;
            int m = e >> 5, k = e & 31;
            int kp = kk + k;
            int c = kp / V_DIM, v = kp % V_DIM;
            armg[i] = qk[qbase + (long)c * L_DIM + (t0 + m) * V_DIM + v];
            brmg[i] = qk[kbase + (long)c * L_DIM + (q0 + m) * V_DIM + v];
        }
    };
    auto storeLds = [&](int p){
        #pragma unroll
        for (int i = 0; i < 16; ++i){
            int e = lane + i * 32;
            int m = e >> 5, k = e & 31;
            As[p][m][k] = armg[i];
            Bs[p][m][k] = brmg[i];
        }
    };

    v8f acc = {};
    const int NC = (CI_DIM * V_DIM) / 32;    // 50
    loadRegs(0);
    storeLds(0);
    int p = 0;
    for (int c = 0; c < NC; ++c){
        __syncthreads();
        if (c + 1 < NC) loadRegs((c + 1) * 32);
        v16bf a, b;
        #pragma unroll
        for (int j = 0; j < 8; ++j){
            int ka = a_kidx(j, g), kb = b_kidx(j, g);
            a[2*j] = As[p][row][ka];  a[2*j+1] = As[p][row][ka + 1];
            b[2*j] = Bs[p][row][kb];  b[2*j+1] = Bs[p][row][kb + 1];
        }
        acc = __builtin_amdgcn_wmma_f32_16x16x32_bf16(false, a, false, b,
                                                      (short)0, acc, false, false);
        if (c + 1 < NC) storeLds(p ^ 1);
        p ^= 1;
    }

    const int q = q0 + row;
    #pragma unroll
    for (int r = 0; r < 8; ++r){
        int t = t0 + r + 8*g;
        float v = tanhf(acc[r] * (1.f / (CI_DIM * V_DIM))) * alphat[s]
                + att0[((long)s * T_DIM + t) * T_DIM + q];
        att[(((long)n * S_DIM + s) * T_DIM + t) * T_DIM + q] = v;
    }
}

// ---------------------------------------------------------------------------
// Temporal apply — biggest GEMM (6400 x 400 x 400 per (n,s)) — WMMA.
// z[n][s*C+c][q][v] = sum_t y[n][c][t][v] * att_t[n][s][t][q]
// 4 waves/block share the attention (B) tile; double-buffered.
// ---------------------------------------------------------------------------
__global__ __launch_bounds__(128)
void temporal_apply_kernel(const float* __restrict__ y, const float* __restrict__ att,
                           __bf16* __restrict__ z)
{
    __shared__ __bf16 As[2][64][34];
    __shared__ __bf16 Bs[2][16][34];
    const int tid = threadIdx.x;
    const int lane = tid & 31, wave = tid >> 5;
    const int q0 = blockIdx.x * 16;
    const int m0 = blockIdx.y * 64;       // over c*V + v (M = 6400)
    const int n = blockIdx.z / S_DIM, s = blockIdx.z % S_DIM;
    const long ybase = (long)n * C_DIM * L_DIM;
    const long abase = ((long)n * S_DIM + s) * T_DIM * T_DIM;
    const int row = lane & 15, g = lane >> 4;

    float armg[16], brmg[4];
    auto loadRegs = [&](int kk){
        #pragma unroll
        for (int i = 0; i < 16; ++i){          // A: 64 x 32
            int e = tid + i * 128;
            int m = e >> 5, k = e & 31;
            int t = kk + k;
            float av = 0.f;
            if (t < T_DIM){
                int mp = m0 + m;
                int cc = mp / V_DIM, v = mp % V_DIM;
                av = y[ybase + (long)cc * L_DIM + t * V_DIM + v];
            }
            armg[i] = av;
        }
        #pragma unroll
        for (int i = 0; i < 4; ++i){           // B: 32 x 16 (q cols)
            int e = tid + i * 128;
            int m = e >> 5, k = e & 31;
            int t = kk + k;
            brmg[i] = (t < T_DIM) ? att[abase + (long)t * T_DIM + q0 + m] : 0.f;
        }
    };
    auto storeLds = [&](int p){
        #pragma unroll
        for (int i = 0; i < 16; ++i){
            int e = tid + i * 128;
            int m = e >> 5, k = e & 31;
            As[p][m][k] = (__bf16)armg[i];
        }
        #pragma unroll
        for (int i = 0; i < 4; ++i){
            int e = tid + i * 128;
            int m = e >> 5, k = e & 31;
            Bs[p][m][k] = (__bf16)brmg[i];
        }
    };

    v8f acc = {};
    const int NC = 13;                         // 13 chunks of 32 cover T=400
    loadRegs(0);
    storeLds(0);
    int p = 0;
    for (int c = 0; c < NC; ++c){
        __syncthreads();
        if (c + 1 < NC) loadRegs((c + 1) * 32);
        v16bf a, b;
        #pragma unroll
        for (int j = 0; j < 8; ++j){
            int ka = a_kidx(j, g), kb = b_kidx(j, g);
            a[2*j] = As[p][wave*16 + row][ka];  a[2*j+1] = As[p][wave*16 + row][ka + 1];
            b[2*j] = Bs[p][row][kb];            b[2*j+1] = Bs[p][row][kb + 1];
        }
        acc = __builtin_amdgcn_wmma_f32_16x16x32_bf16(false, a, false, b,
                                                      (short)0, acc, false, false);
        if (c + 1 < NC) storeLds(p ^ 1);
        p ^= 1;
    }

    const int q = q0 + row;
    #pragma unroll
    for (int r = 0; r < 8; ++r){
        int mp = m0 + wave*16 + r + 8*g;
        int c = mp / V_DIM, v = mp % V_DIM;
        z[((long)n * ZC + s * C_DIM + c) * L_DIM + q * V_DIM + v] = (__bf16)acc[r];
    }
}

// ---------------------------------------------------------------------------
extern "C" void kernel_launch(void* const* d_in, const int* in_sizes, int n_in,
                              void* d_out, int out_size, void* d_ws, size_t ws_size,
                              hipStream_t stream)
{
    (void)in_sizes; (void)n_in; (void)out_size; (void)ws_size;
    const float* x            = (const float*)d_in[0];
    const float* x1           = (const float*)d_in[1];
    const float* pe_s         = (const float*)d_in[2];
    const float* pe_t         = (const float*)d_in[3];
    const float* in_nets_W    = (const float*)d_in[4];
    const float* in_nets_b    = (const float*)d_in[5];
    const float* alphas       = (const float*)d_in[6];
    const float* att0s        = (const float*)d_in[7];
    const float* out_nets_W   = (const float*)d_in[8];
    const float* out_nets_b   = (const float*)d_in[9];
    const float* out_nets_g   = (const float*)d_in[10];
    const float* out_nets_be  = (const float*)d_in[11];
    const float* ff_nets_W    = (const float*)d_in[12];
    const float* ff_nets_b    = (const float*)d_in[13];
    const float* ff_nets_g    = (const float*)d_in[14];
    const float* ff_nets_be   = (const float*)d_in[15];
    const float* in_nett_W    = (const float*)d_in[16];
    const float* in_nett_b    = (const float*)d_in[17];
    const float* alphat       = (const float*)d_in[18];
    const float* att0t        = (const float*)d_in[19];
    const float* out_nett_W   = (const float*)d_in[20];
    const float* out_nett_b   = (const float*)d_in[21];
    const float* out_nett_g   = (const float*)d_in[22];
    const float* out_nett_be  = (const float*)d_in[23];
    const float* ff_nett_W    = (const float*)d_in[24];
    const float* ff_nett_b    = (const float*)d_in[25];
    const float* ff_nett_g    = (const float*)d_in[26];
    const float* ff_nett_be   = (const float*)d_in[27];
    // d_in[28] = 'case' (unused by reference math)

    const size_t NEL = (size_t)NB * C_DIM * L_DIM;      // 10,240,000
    float* ws    = (float*)d_ws;
    float*  Y    = ws;                                   // f32 sp       10.24M
    __bf16* Yb   = (__bf16*)(Y + NEL);                   // bf16 y       (5.12M f)
    __bf16* QKb  = (__bf16*)(Y + NEL + NEL/2);           // bf16 qk      (7.68M f)
    __bf16* Zb   = QKb + (size_t)NB * QKC * L_DIM;       // bf16 z       (15.36M f)
    __bf16* Hb   = Zb + (size_t)NB * ZC * L_DIM;         // bf16 h       (5.12M f)
    float*  ATTS = (float*)(Hb + NEL);                   // 7.5K
    float*  ATTT = ATTS + 8192;                          // 1.92M
    __bf16* Wb0  = (__bf16*)(ATTT + (size_t)NB*S_DIM*T_DIM*T_DIM);
    __bf16* Wb_ins  = Wb0;                               // 384*256
    __bf16* Wb_outs = Wb_ins  + QKC*C_DIM;               // 256*768
    __bf16* Wb_ffs  = Wb_outs + C_DIM*ZC;                // 256*256
    __bf16* Wb_int  = Wb_ffs  + C_DIM*C_DIM;
    __bf16* Wb_outt = Wb_int  + QKC*C_DIM;
    __bf16* Wb_fft  = Wb_outt + C_DIM*ZC;

    // one-time weight conversions (cheap, ~1.4 MB bf16 total)
    cvt_bf16_kernel<<<(QKC*C_DIM+255)/256, 256, 0, stream>>>(in_nets_W, Wb_ins, QKC*C_DIM);
    cvt_bf16_kernel<<<(C_DIM*ZC +255)/256, 256, 0, stream>>>(out_nets_W, Wb_outs, C_DIM*ZC);
    cvt_bf16_kernel<<<(C_DIM*C_DIM+255)/256,256,0, stream>>>(ff_nets_W, Wb_ffs, C_DIM*C_DIM);
    cvt_bf16_kernel<<<(QKC*C_DIM+255)/256, 256, 0, stream>>>(in_nett_W, Wb_int, QKC*C_DIM);
    cvt_bf16_kernel<<<(C_DIM*ZC +255)/256, 256, 0, stream>>>(out_nett_W, Wb_outt, C_DIM*ZC);
    cvt_bf16_kernel<<<(C_DIM*C_DIM+255)/256,256,0, stream>>>(ff_nett_W, Wb_fft, C_DIM*C_DIM);

    const float* xins[2] = { x, x1 };
    for (int chain = 0; chain < 2; ++chain){
        const float* xin = xins[chain];
        float* outp = (float*)d_out + (size_t)chain * NEL;

        // ---- spatial ----
        add_pe_kernel<<<40000, 256, 0, stream>>>(xin, pe_s, Yb, (int)NEL);
        conv_gemm_kernel<2><<<dim3(625, 3, NB), 128, 0, stream>>>(        // O=384,K=256
            Wb_ins, in_nets_b, Yb, QKb, QKC, C_DIM, nullptr, nullptr, nullptr, 0, 1);
        spatial_att_kernel<<<dim3(625, NB * S_DIM), 256, 0, stream>>>(QKb, alphas, att0s, ATTS);
        spatial_apply_kernel<<<120000, 256, 0, stream>>>(xin, ATTS, Zb);
        conv_gemm_kernel<4><<<dim3(625, 1, NB), 128, 0, stream>>>(        // O=256,K=768
            Wb_outs, out_nets_b, Zb, Hb, C_DIM, ZC, xin, out_nets_g, out_nets_be, 1, 1);
        conv_gemm_kernel<4><<<dim3(625, 1, NB), 128, 0, stream>>>(        // O=256,K=256
            Wb_ffs, ff_nets_b, Hb, Y, C_DIM, C_DIM, xin, ff_nets_g, ff_nets_be, 1, 0);
        // Y now holds sp = spatial(xin) in f32

        // ---- temporal ----
        add_pe_kernel<<<40000, 256, 0, stream>>>(Y, pe_t, Yb, (int)NEL);
        conv_gemm_kernel<2><<<dim3(625, 3, NB), 128, 0, stream>>>(
            Wb_int, in_nett_b, Yb, QKb, QKC, C_DIM, nullptr, nullptr, nullptr, 0, 1);
        temporal_att_kernel<<<dim3(25, 25, NB * S_DIM), 32, 0, stream>>>(QKb, alphat, att0t, ATTT);
        temporal_apply_kernel<<<dim3(25, 100, NB * S_DIM), 128, 0, stream>>>(Y, ATTT, Zb);
        conv_gemm_kernel<4><<<dim3(625, 1, NB), 128, 0, stream>>>(
            Wb_outt, out_nett_b, Zb, Hb, C_DIM, ZC, Y, out_nett_g, out_nett_be, 1, 1);
        conv_gemm_kernel<4><<<dim3(625, 1, NB), 128, 0, stream>>>(
            Wb_fft, ff_nett_b, Hb, outp, C_DIM, C_DIM, Y, ff_nett_g, ff_nett_be, 1, 0);
    }
}
